// RelationalMemoryCore_39316130627979
// MI455X (gfx1250) — matrix operand order repaired and checked by reading, not compiled, not run on hardware
//
#include <hip/hip_runtime.h>
#include <hip/hip_bf16.h>

// ---------------------------------------------------------------------------
// Types for CDNA5 WMMA (wave32): v_wmma_f32_16x16x32_bf16
// ---------------------------------------------------------------------------
typedef __attribute__((ext_vector_type(16))) __bf16 v16bf;
typedef __attribute__((ext_vector_type(8)))  float  v8f;
typedef __attribute__((ext_vector_type(4)))  unsigned int u32x4;
typedef __attribute__((ext_vector_type(8)))  int i32x8;
typedef __attribute__((ext_vector_type(4)))  int i32x4;

union FragBF { v16bf v; __bf16 e[16]; uint4 q[2]; };

static __device__ __forceinline__ __bf16 f2bf(float f) {
    union { float f; unsigned u; } x; x.f = f;
    unsigned r = x.u + 0x7FFFu + ((x.u >> 16) & 1u);   // round-to-nearest-even
    union { unsigned short s; __bf16 b; } y; y.s = (unsigned short)(r >> 16);
    return y.b;
}

// ---------------------------------------------------------------------------
// Tensor Data Mover: 2D tile (tile_d1 rows x tile_d0 elems of 2B) from global
// row-major [tensor_d1][tensor_d0] (row stride d0_stride elems) into LDS,
// with LDS padding of 4 DWORDs after every 16 DWORDs (64B row -> 80B stride,
// i.e. 40 bf16 = LDSTRIDE). Descriptor layout per CDNA5 ISA ch.8 (D#).
// ---------------------------------------------------------------------------
#if __has_include(<hip/amd_detail/amd_gfx1250_TDM.h>)
#define TDM_CALL(g0, g1, g2, g3) \
    __builtin_amdgcn_tensor_load_to_lds(g0, g1, g2, g3, (i32x8){0,0,0,0,0,0,0,0}, 0)
#else
#define TDM_CALL(g0, g1, g2, g3) \
    __builtin_amdgcn_tensor_load_to_lds(g0, g1, g2, g3, 0)
#endif

static __device__ __forceinline__ void tdm_load_tile_2d(unsigned lds_addr,
                                                        const void* gaddr,
                                                        unsigned tensor_d0,
                                                        unsigned tensor_d1,
                                                        unsigned d0_stride,
                                                        unsigned tile_d0,
                                                        unsigned tile_d1) {
    unsigned long long ga = (unsigned long long)(uintptr_t)gaddr;
    u32x4 g0;
    g0[0] = 1u;                                       // count=1, user descriptor
    g0[1] = lds_addr;                                 // lds_addr [63:32]
    g0[2] = (unsigned)(ga & 0xFFFFFFFFu);             // global_addr [95:64]
    g0[3] = (unsigned)((ga >> 32) & 0x1FFFFFFu)       // global_addr [120:96]
          | (2u << 30);                               // type=2 ("image")
    i32x8 g1;
    // word0: workgroup_mask=0 | data_size=1(2B)<<16 | pad_enable<<20 |
    //        pad_interval=3 (16 DWORDs)<<22 | pad_amount=3 (4 DWORDs)<<25
    g1[0] = (int)((1u << 16) | (1u << 20) | (3u << 22) | (3u << 25));
    g1[1] = (int)((tensor_d0 & 0xFFFFu) << 16);                           // dim0[15:0] @ bit48
    g1[2] = (int)(((tensor_d0 >> 16) & 0xFFFFu) | ((tensor_d1 & 0xFFFFu) << 16));
    g1[3] = (int)(((tensor_d1 >> 16) & 0xFFFFu) | ((tile_d0 & 0xFFFFu) << 16));
    g1[4] = (int)(tile_d1 & 0xFFFFu);                                     // tile_dim1; tile_dim2=0
    g1[5] = (int)d0_stride;                                               // dim0_stride[31:0]
    g1[6] = 0;                                                            // dim0_stride[47:32], dim1_stride lo
    g1[7] = 0;                                                            // dim1_stride hi
    i32x4 z4 = {0, 0, 0, 0};
    TDM_CALL(g0, g1, z4, z4);
}

static __device__ __forceinline__ unsigned lds_off(const void* p) {
    return (unsigned)(uintptr_t)p;   // flat LDS address: offset in addr[31:0]
}

// ---------------------------------------------------------------------------
// f32 -> bf16 conversion (no transpose) for activations
// ---------------------------------------------------------------------------
__global__ __launch_bounds__(256) void cvt_bf16_kernel(const float* __restrict__ src,
                                                       __bf16* __restrict__ dst, int n) {
    int i = blockIdx.x * 256 + threadIdx.x;
    if (i < n) dst[i] = f2bf(src[i]);
}

// ---------------------------------------------------------------------------
// f32 [R][C] -> bf16 [C][R] tiled transpose (weights become [N][K] so the
// WMMA B-fragment is contiguous per lane in LDS -> ds_load_b128)
// ---------------------------------------------------------------------------
__global__ __launch_bounds__(256) void cvt_transpose_kernel(const float* __restrict__ src,
                                                            __bf16* __restrict__ dst,
                                                            int R, int C) {
    __shared__ float tile[32][33];
    int c0 = blockIdx.x * 32;
    int r0 = blockIdx.y * 32;
    int tx = threadIdx.x;   // 0..31
    int ty = threadIdx.y;   // 0..7
#pragma unroll
    for (int i = 0; i < 4; ++i) {
        int r = ty + i * 8;
        tile[r][tx] = src[(size_t)(r0 + r) * C + c0 + tx];
    }
    __syncthreads();
#pragma unroll
    for (int i = 0; i < 4; ++i) {
        int r = ty + i * 8;   // row in dst tile == col in src
        dst[(size_t)(c0 + r) * R + r0 + tx] = f2bf(tile[tx][r]);
    }
}

// ---------------------------------------------------------------------------
// Build m = concat(memory, inp) in f32 + bf16, plus bf16 copies of memory / inp
// ---------------------------------------------------------------------------
__global__ __launch_bounds__(256) void pack_m_kernel(const float* __restrict__ memory,
                                                     const float* __restrict__ inp,
                                                     float* __restrict__ mF,
                                                     __bf16* __restrict__ mB,
                                                     __bf16* __restrict__ memB,
                                                     __bf16* __restrict__ inpB) {
    const int TOT = 4096 * 9 * 1024;
    int i = blockIdx.x * 256 + threadIdx.x;
    if (i >= TOT) return;
    int j    = i % 1024;
    int slot = (i / 1024) % 9;
    int b    = i / (9 * 1024);
    float v;
    if (slot < 8) {
        size_t mi = ((size_t)b * 8 + slot) * 1024 + j;
        v = memory[mi];
        memB[mi] = f2bf(v);
    } else {
        size_t ii = (size_t)b * 1024 + j;
        v = inp[ii];
        inpB[ii] = f2bf(v);
    }
    mF[i] = v;
    mB[i] = f2bf(v);
}

// ---------------------------------------------------------------------------
// bf16 WMMA GEMM with TDM-staged, double-buffered LDS tiles.
//   C[M,N] = A[M,K] @ B[K,N] (+bias) (+relu) -> f32 or bf16
//   A  : row-major [M][K] bf16 ; Bt : TRANSPOSED weights [N][K] bf16
// Block: 256 threads = 8 waves; macro-tile 128x128; K-step 32.
// Wave w computes rows [16w,16w+16) x 128 cols -> 8 f32 accumulators.
// Staging: wave 0 issues TENSOR_LOAD_TO_LDS for tile t+1 while all waves
// run WMMAs on tile t; sync = s_wait_tensorcnt + workgroup barriers.
// TDM pad config reproduces LDSTRIDE=40 (80B row stride, conflict-free).
// ---------------------------------------------------------------------------
#define LDSTRIDE 40   // 32 data + 8 pad bf16 per row

template <int OUT_BF16, int RELU>
__global__ __launch_bounds__(256) void gemm_bf16_kernel(const __bf16* __restrict__ A,
                                                        const __bf16* __restrict__ Bt,
                                                        const float* __restrict__ bias,
                                                        float* __restrict__ Cf,
                                                        __bf16* __restrict__ Cb,
                                                        int M, int N, int K) {
    __shared__ __attribute__((aligned(16))) __bf16 As[2][128 * LDSTRIDE];
    __shared__ __attribute__((aligned(16))) __bf16 Bs[2][128 * LDSTRIDE];

    const int tid  = threadIdx.x;
    const int lane = tid & 31;
    const int wave = tid >> 5;      // 0..7
    const int lr   = lane & 15;
    const int hi   = lane >> 4;
    const int tileM = blockIdx.y * 128;
    const int tileN = blockIdx.x * 128;

    v8f zero = {0.f, 0.f, 0.f, 0.f, 0.f, 0.f, 0.f, 0.f};
    v8f acc[8];
#pragma unroll
    for (int n = 0; n < 8; ++n) acc[n] = zero;

    const bool issuer = (wave == 0);
    const int nsteps = K / 32;

    const __bf16* aBase = A  + (size_t)tileM * K;
    const __bf16* bBase = Bt + (size_t)tileN * K;

    if (issuer) {   // prologue: DMA tile 0 (A: 128x32, B: 128x32)
        tdm_load_tile_2d(lds_off(&As[0][0]), aBase, (unsigned)K, (unsigned)M,
                         (unsigned)K, 32u, 128u);
        tdm_load_tile_2d(lds_off(&Bs[0][0]), bBase, (unsigned)K, (unsigned)N,
                         (unsigned)K, 32u, 128u);
    }

    for (int t = 0; t < nsteps; ++t) {
        const int cur = t & 1;
        const int nxt = cur ^ 1;
        if (issuer) {
            if (t + 1 < nsteps) {
                const int k1 = (t + 1) * 32;
                tdm_load_tile_2d(lds_off(&As[nxt][0]), aBase + k1, (unsigned)K,
                                 (unsigned)M, (unsigned)K, 32u, 128u);
                tdm_load_tile_2d(lds_off(&Bs[nxt][0]), bBase + k1, (unsigned)K,
                                 (unsigned)N, (unsigned)K, 32u, 128u);
                __builtin_amdgcn_s_wait_tensorcnt(2);   // tile t's pair complete
            } else {
                __builtin_amdgcn_s_wait_tensorcnt(0);
            }
        }
        __syncthreads();   // publish buffer `cur` to all waves

        // A fragment: row = wave*16 + lr ; chunks at k = hi*8 and 16 + hi*8
        FragBF a;
        a.q[0] = *reinterpret_cast<const uint4*>(&As[cur][(wave * 16 + lr) * LDSTRIDE + hi * 8]);
        a.q[1] = *reinterpret_cast<const uint4*>(&As[cur][(wave * 16 + lr) * LDSTRIDE + 16 + hi * 8]);

#pragma unroll
        for (int n = 0; n < 8; ++n) {
            // B fragment: col = n*16 + lr ; chunks at k = hi*8 and 16 + hi*8
            FragBF bf;
            bf.q[0] = *reinterpret_cast<const uint4*>(&Bs[cur][(n * 16 + lr) * LDSTRIDE + hi * 8]);
            bf.q[1] = *reinterpret_cast<const uint4*>(&Bs[cur][(n * 16 + lr) * LDSTRIDE + 16 + hi * 8]);
            acc[n] = __builtin_amdgcn_wmma_f32_16x16x32_bf16(
                false, a.v, false, bf.v, (short)0, acc[n], false, false);
        }
        __syncthreads();   // all reads of `cur` done before it is re-DMAed
    }

    // Epilogue. C/D layout: VGPR i: lane<16 -> (M=i, N=lane); lane>=16 -> (M=i+8, N=lane-16)
#pragma unroll
    for (int n = 0; n < 8; ++n) {
        int col  = tileN + n * 16 + lr;
        float bv = bias ? bias[col] : 0.0f;
#pragma unroll
        for (int i = 0; i < 8; ++i) {
            int row = tileM + wave * 16 + hi * 8 + i;
            float val = acc[n][i] + bv;
            if (RELU) val = val > 0.0f ? val : 0.0f;
            if (OUT_BF16) Cb[(size_t)row * N + col] = f2bf(val);
            else          Cf[(size_t)row * N + col] = val;
        }
    }
}

// ---------------------------------------------------------------------------
// In-place row LayerNorm (row length = rowlen), one 256-thread block per row
// ---------------------------------------------------------------------------
__global__ __launch_bounds__(256) void ln_inplace_kernel(float* __restrict__ x,
                                                         const float* __restrict__ g,
                                                         const float* __restrict__ beta,
                                                         int rowlen) {
    __shared__ float rs[256], rq[256];
    size_t base = (size_t)blockIdx.x * rowlen;
    float s = 0.f, q = 0.f;
    for (int j = threadIdx.x; j < rowlen; j += 256) {
        float v = x[base + j];
        s += v; q += v * v;
    }
    rs[threadIdx.x] = s; rq[threadIdx.x] = q;
    __syncthreads();
    for (int st = 128; st > 0; st >>= 1) {
        if ((int)threadIdx.x < st) {
            rs[threadIdx.x] += rs[threadIdx.x + st];
            rq[threadIdx.x] += rq[threadIdx.x + st];
        }
        __syncthreads();
    }
    float mean = rs[0] / rowlen;
    float var  = rq[0] / rowlen - mean * mean;
    float rstd = rsqrtf(var + 1e-5f);
    for (int j = threadIdx.x; j < rowlen; j += 256) {
        float v = x[base + j];
        x[base + j] = (v - mean) * rstd * g[j] + beta[j];
    }
}

// ---------------------------------------------------------------------------
// LayerNorm(a + r) -> f32 (and optional bf16), one block per row of `rowlen`
// ---------------------------------------------------------------------------
__global__ __launch_bounds__(256) void ln_res_kernel(const float* __restrict__ a,
                                                     const float* __restrict__ r,
                                                     const float* __restrict__ g,
                                                     const float* __restrict__ beta,
                                                     float* __restrict__ outf,
                                                     __bf16* __restrict__ outb,
                                                     int rowlen) {
    __shared__ float rs[256], rq[256];
    size_t base = (size_t)blockIdx.x * rowlen;
    float s = 0.f, q = 0.f;
    for (int j = threadIdx.x; j < rowlen; j += 256) {
        float v = a[base + j] + r[base + j];
        s += v; q += v * v;
    }
    rs[threadIdx.x] = s; rq[threadIdx.x] = q;
    __syncthreads();
    for (int st = 128; st > 0; st >>= 1) {
        if ((int)threadIdx.x < st) {
            rs[threadIdx.x] += rs[threadIdx.x + st];
            rq[threadIdx.x] += rq[threadIdx.x + st];
        }
        __syncthreads();
    }
    float mean = rs[0] / rowlen;
    float var  = rq[0] / rowlen - mean * mean;
    float rstd = rsqrtf(var + 1e-5f);
    for (int j = threadIdx.x; j < rowlen; j += 256) {
        float v = (a[base + j] + r[base + j] - mean) * rstd * g[j] + beta[j];
        outf[base + j] = v;
        if (outb) outb[base + j] = f2bf(v);
    }
}

// ---------------------------------------------------------------------------
// Tiny 9x9 per-head attention. One thread per (b, head, n) output row.
// qkv layout per row of 3072: [head*192 + {q:0, k:64, v:128} + d]
// ---------------------------------------------------------------------------
__global__ __launch_bounds__(256) void attn_kernel(const float* __restrict__ qkv,
                                                   float* __restrict__ att) {
    const int TOT = 4096 * 16 * 9;
    int idx = blockIdx.x * 256 + threadIdx.x;
    if (idx >= TOT) return;
    int n = idx % 9;
    int h = (idx / 9) % 16;
    int b = idx / (9 * 16);
    const float* base = qkv + (size_t)b * 9 * 3072 + h * 192;

    float qv[64];
#pragma unroll
    for (int d = 0; d < 64; ++d) qv[d] = base[(size_t)n * 3072 + d] * 0.125f;  // 1/sqrt(64)

    float p[9];
    float mx = -1e30f;
    for (int m = 0; m < 9; ++m) {
        const float* krow = base + (size_t)m * 3072 + 64;
        float dot = 0.f;
#pragma unroll
        for (int d = 0; d < 64; ++d) dot += qv[d] * krow[d];
        p[m] = dot;
        mx = fmaxf(mx, dot);
    }
    float sum = 0.f;
    for (int m = 0; m < 9; ++m) { p[m] = __expf(p[m] - mx); sum += p[m]; }
    float inv = 1.f / sum;

    float out[64];
#pragma unroll
    for (int d = 0; d < 64; ++d) out[d] = 0.f;
    for (int m = 0; m < 9; ++m) {
        const float* vrow = base + (size_t)m * 3072 + 128;
        float w = p[m] * inv;
#pragma unroll
        for (int d = 0; d < 64; ++d) out[d] += w * vrow[d];
    }
    float* orow = att + ((size_t)b * 9 + n) * 1024 + h * 64;
#pragma unroll
    for (int d = 0; d < 64; ++d) orow[d] = out[d];
}

// ---------------------------------------------------------------------------
// Final gating: next = sigmoid(i)*tanh(m2) + sigmoid(f)*memory
// Writes both tuple outputs (output, next_memory) — identical data.
// ---------------------------------------------------------------------------
__global__ __launch_bounds__(256) void gates_kernel(const float* __restrict__ g1,
                                                    const float* __restrict__ g2,
                                                    const float* __restrict__ ig_b,
                                                    const float* __restrict__ mg_b,
                                                    const float* __restrict__ m2,
                                                    const float* __restrict__ memory,
                                                    const float* __restrict__ input_bias,
                                                    const float* __restrict__ forget_bias,
                                                    float* __restrict__ out) {
    const int TOT = 4096 * 8 * 1024;
    int idx = blockIdx.x * 256 + threadIdx.x;
    if (idx >= TOT) return;
    int j = idx % 1024;
    int s = (idx / 1024) % 8;
    int b = idx / (8 * 1024);
    size_t r2 = ((size_t)b * 8 + s) * 2048;
    size_t r1 = (size_t)b * 2048;
    float gi = g2[r2 + j]        + mg_b[j]        + g1[r1 + j]        + ig_b[j]        + input_bias[0];
    float gf = g2[r2 + 1024 + j] + mg_b[1024 + j] + g1[r1 + 1024 + j] + ig_b[1024 + j] + forget_bias[0];
    float ig = 1.f / (1.f + __expf(-gi));
    float fg = 1.f / (1.f + __expf(-gf));
    float nm = m2[((size_t)b * 9 + s) * 1024 + j];
    float res = ig * tanhf(nm) + fg * memory[idx];
    out[idx] = res;                     // output
    out[(size_t)TOT + idx] = res;       // next_memory
}

// ---------------------------------------------------------------------------
// Host orchestration
// ---------------------------------------------------------------------------
extern "C" void kernel_launch(void* const* d_in, const int* in_sizes, int n_in,
                              void* d_out, int out_size, void* d_ws, size_t ws_size,
                              hipStream_t stream) {
    (void)in_sizes; (void)n_in; (void)out_size; (void)ws_size;

    const float* input_vec   = (const float*)d_in[0];
    const float* memory      = (const float*)d_in[1];
    const float* qkv_w       = (const float*)d_in[2];
    const float* qkv_b       = (const float*)d_in[3];
    const float* qkvn_g      = (const float*)d_in[4];
    const float* qkvn_b      = (const float*)d_in[5];
    const float* mlp1_w      = (const float*)d_in[6];
    const float* mlp1_b      = (const float*)d_in[7];
    const float* mlp2_w      = (const float*)d_in[8];
    const float* mlp2_b      = (const float*)d_in[9];
    const float* n1_g        = (const float*)d_in[10];
    const float* n1_b        = (const float*)d_in[11];
    const float* n2_g        = (const float*)d_in[12];
    const float* n2_b        = (const float*)d_in[13];
    const float* inp_w       = (const float*)d_in[14];
    const float* inp_b       = (const float*)d_in[15];
    const float* ig_w        = (const float*)d_in[16];
    const float* ig_b        = (const float*)d_in[17];
    const float* mg_w        = (const float*)d_in[18];
    const float* mg_b        = (const float*)d_in[19];
    const float* forget_bias = (const float*)d_in[20];
    const float* input_bias  = (const float*)d_in[21];

    const int Bn = 4096, MEMd = 1024, QKVd = 3072, Gd = 2048;
    const int ROWS = Bn * 9;   // 36864
    const int MR   = Bn * 8;   // 32768

    char* ws = (char*)d_ws;
    size_t off = 0;
    auto take = [&](size_t bytes) -> void* {
        void* p = ws + off;
        off += (bytes + 255) & ~(size_t)255;
        return p;
    };

    // All weight buffers hold TRANSPOSED bf16 weights: [N][K]
    __bf16* w_inp  = (__bf16*)take((size_t)MEMd * MEMd * 2);
    __bf16* w_qkv  = (__bf16*)take((size_t)MEMd * QKVd * 2);
    __bf16* w_mlp1 = (__bf16*)take((size_t)MEMd * MEMd * 2);
    __bf16* w_mlp2 = (__bf16*)take((size_t)MEMd * MEMd * 2);
    __bf16* w_ig   = (__bf16*)take((size_t)MEMd * Gd * 2);
    __bf16* w_mg   = (__bf16*)take((size_t)MEMd * Gd * 2);
    __bf16* ivB    = (__bf16*)take((size_t)Bn * MEMd * 2);
    float*  inpF   = (float*) take((size_t)Bn * MEMd * 4);
    float*  mF     = (float*) take((size_t)ROWS * MEMd * 4);   // m, later m2
    __bf16* mB     = (__bf16*)take((size_t)ROWS * MEMd * 2);   // m_bf16, later relu hidden
    __bf16* memB   = (__bf16*)take((size_t)MR * MEMd * 2);
    __bf16* inpB   = (__bf16*)take((size_t)Bn * MEMd * 2);
    float*  qkvF   = (float*) take((size_t)ROWS * QKVd * 4);   // qkv, later g1+g2
    float*  attF   = (float*) take((size_t)ROWS * MEMd * 4);   // attended, later mlp
    float*  m1F    = (float*) take((size_t)ROWS * MEMd * 4);
    __bf16* m1B    = (__bf16*)take((size_t)ROWS * MEMd * 2);
    float*  g1 = qkvF;                         // [Bn, 2048]
    float*  g2 = qkvF + (size_t)Bn * Gd;       // [MR, 2048]

    auto cgrid = [](size_t n) { return dim3((unsigned)((n + 255) / 256)); };
    dim3 tblk(32, 8);
    dim3 gblk(256);

    // 1. weight conversion + transpose to bf16 [N][K]; input_vec plain convert
    cvt_transpose_kernel<<<dim3(MEMd / 32, MEMd / 32), tblk, 0, stream>>>(inp_w,  w_inp,  MEMd, MEMd);
    cvt_transpose_kernel<<<dim3(QKVd / 32, MEMd / 32), tblk, 0, stream>>>(qkv_w,  w_qkv,  MEMd, QKVd);
    cvt_transpose_kernel<<<dim3(MEMd / 32, MEMd / 32), tblk, 0, stream>>>(mlp1_w, w_mlp1, MEMd, MEMd);
    cvt_transpose_kernel<<<dim3(MEMd / 32, MEMd / 32), tblk, 0, stream>>>(mlp2_w, w_mlp2, MEMd, MEMd);
    cvt_transpose_kernel<<<dim3(Gd / 32,   MEMd / 32), tblk, 0, stream>>>(ig_w,   w_ig,   MEMd, Gd);
    cvt_transpose_kernel<<<dim3(Gd / 32,   MEMd / 32), tblk, 0, stream>>>(mg_w,   w_mg,   MEMd, Gd);
    cvt_bf16_kernel<<<cgrid((size_t)Bn * MEMd), 256, 0, stream>>>(input_vec, ivB, Bn * MEMd);

    // 2. inp = input_vec @ inp_w + inp_b
    gemm_bf16_kernel<0, 0><<<dim3(MEMd / 128, Bn / 128), gblk, 0, stream>>>(
        ivB, w_inp, inp_b, inpF, nullptr, Bn, MEMd, MEMd);

    // 3. m = concat(memory, inp)
    pack_m_kernel<<<cgrid((size_t)ROWS * MEMd), 256, 0, stream>>>(memory, inpF, mF, mB, memB, inpB);

    // 4. qkv = m @ qkv_w + qkv_b ; 5. LayerNorm over 3072
    gemm_bf16_kernel<0, 0><<<dim3(QKVd / 128, ROWS / 128), gblk, 0, stream>>>(
        mB, w_qkv, qkv_b, qkvF, nullptr, ROWS, QKVd, MEMd);
    ln_inplace_kernel<<<dim3(ROWS), 256, 0, stream>>>(qkvF, qkvn_g, qkvn_b, QKVd);

    // 6. attention (9x9 per head)
    attn_kernel<<<cgrid((size_t)Bn * 16 * 9), 256, 0, stream>>>(qkvF, attF);

    // 7. m1 = LN(m + attended)
    ln_res_kernel<<<dim3(ROWS), 256, 0, stream>>>(mF, attF, n1_g, n1_b, m1F, m1B, MEMd);

    // 8. h = relu(m1 @ mlp1_w + mlp1_b)  (bf16 out)
    gemm_bf16_kernel<1, 1><<<dim3(MEMd / 128, ROWS / 128), gblk, 0, stream>>>(
        m1B, w_mlp1, mlp1_b, nullptr, mB, ROWS, MEMd, MEMd);

    // 9. mlp = h @ mlp2_w + mlp2_b
    gemm_bf16_kernel<0, 0><<<dim3(MEMd / 128, ROWS / 128), gblk, 0, stream>>>(
        mB, w_mlp2, mlp2_b, attF, nullptr, ROWS, MEMd, MEMd);

    // 10. m2 = LN(m1 + mlp)
    ln_res_kernel<<<dim3(ROWS), 256, 0, stream>>>(m1F, attF, n2_g, n2_b, mF, nullptr, MEMd);

    // 11/12. gate GEMMs: g1 = inp @ ig_w ; g2 = memory @ mg_w
    gemm_bf16_kernel<0, 0><<<dim3(Gd / 128, Bn / 128), gblk, 0, stream>>>(
        inpB, w_ig, nullptr, g1, nullptr, Bn, Gd, MEMd);
    gemm_bf16_kernel<0, 0><<<dim3(Gd / 128, MR / 128), gblk, 0, stream>>>(
        memB, w_mg, nullptr, g2, nullptr, MR, Gd, MEMd);

    // 13. gating + both tuple outputs
    gates_kernel<<<cgrid((size_t)Bn * 8 * MEMd), 256, 0, stream>>>(
        g1, g2, ig_b, mg_b, mF, memory, input_bias, forget_bias, (float*)d_out);
}
